// q_82059645157709
// MI455X (gfx1250) — matrix operand order repaired and checked by reading, not compiled, not run on hardware
//
#include <hip/hip_runtime.h>
#include <hip/hip_bf16.h>

typedef __attribute__((ext_vector_type(16))) _Float16 v16h;
typedef __attribute__((ext_vector_type(8)))  float    v8f;

#define SIGMA 1.0f
#define NCENT 16

// One wave handles 32 elements per iteration with ONE v_wmma_f32_16x16x32_f16:
//   B (32x16 f16): lane L contributes its own element's 16 weights w_j = exp(-|x_L - c_j|).
//     With the contiguous K order used here (K = j + 16*hi), lanes 0-15 fill K=0..15 of column L (elements base+0..15),
//     lanes 16-31 fill K=16..31 of column L-16 (elements base+16..31).
//   A (16x32 f16, loop-invariant): row0 = centers at K<16, row1 = ones at K<16,
//     row2 = centers at K>=16, row3 = ones at K>=16, other rows zero.
//   => D row0 = numerator t(elem n), row1 = denom s(elem n),
//      D row2 = t(elem n+16),        row3 = s(elem n+16)   (all in lanes 0-15).
// soft = t * rcp(s) (fast rcp is fine: forward output == hard to ~1 ulp regardless
// of soft precision, per the straight-through identity (hard-soft)+soft).
// hard tracked per-lane during the exp loop with exact f32 distances.
__global__ __launch_bounds__(256) void softquant_wmma_kernel(
    const float* __restrict__ x, const float* __restrict__ centers,
    float* __restrict__ out, int n)
{
  const int lane   = threadIdx.x & 31;
  const int wave   = (int)((blockIdx.x * blockDim.x + threadIdx.x) >> 5);
  const int nwaves = (int)((gridDim.x * blockDim.x) >> 5);

  // Centers: uniform address -> scalar loads; 16 fp32 regs.
  float c[NCENT];
#pragma unroll
  for (int j = 0; j < NCENT; ++j) c[j] = centers[j];

  // Build loop-invariant A operand (16x32 f16), in contiguous K order (same as B):
  // lane holds row m = lane&15; half i is K = i + 16*hi.
  const int m  = lane & 15;
  const int hi = lane >> 4;
  v16h A;
#pragma unroll
  for (int i = 0; i < 16; ++i) {
    const int k = i + 16 * hi;
    float v = 0.0f;
    if      (m == 0) v = (k < 16)  ? c[k]      : 0.0f;
    else if (m == 1) v = (k < 16)  ? 1.0f      : 0.0f;
    else if (m == 2) v = (k >= 16) ? c[k - 16] : 0.0f;
    else if (m == 3) v = (k >= 16) ? 1.0f      : 0.0f;
    A[i] = (_Float16)v;
  }

  const int ntiles = n >> 5;  // 32 elements per tile; N is a multiple of 32
  for (int t = wave; t < ntiles; t += nwaves) {
    const int base = t << 5;
    const int e    = base + lane;       // always < n since t < ntiles
    const float xv = x[e];

    // Prefetch next grid-stride tile (gfx1250 global_prefetch_b8).
    if (t + nwaves < ntiles)
      __builtin_prefetch(&x[(size_t)(t + nwaves) << 5], 0, 3);

    // 16 exps (trans pipe) + exact nearest-center tracking; build B operand.
    float best = 3.4e38f;
    float bc   = c[0];
    v16h B;
#pragma unroll
    for (int j = 0; j < NCENT; ++j) {
      const float ad = fabsf(xv - c[j]);
      const float w  = __expf(-SIGMA * ad);
      B[j] = (_Float16)w;
      const bool lt = ad < best;       // strict < keeps first min == argmin
      best = lt ? ad : best;
      bc   = lt ? c[j] : bc;
    }

    v8f acc = {0.f, 0.f, 0.f, 0.f, 0.f, 0.f, 0.f, 0.f};
    acc = __builtin_amdgcn_wmma_f32_16x16x32_f16(
        /*neg_a=*/false, A, /*neg_b=*/false, B,
        /*c_mod=*/(short)0, acc, /*reuse_a=*/false, /*reuse_b=*/false);

    // lanes 0-15: acc[0]/acc[1] = soft(base+lane); acc[2]/acc[3] = soft(base+16+lane)
    // Fast reciprocal instead of IEEE divide (see note above).
    const float soft_lo = acc[0] * __builtin_amdgcn_rcpf(acc[1]);
    const float soft_hi = acc[2] * __builtin_amdgcn_rcpf(acc[3]);
    const float sh      = __shfl_xor(soft_hi, 16, 32);
    const float soft    = (lane < 16) ? soft_lo : sh;

    out[e] = (bc - soft) + soft;  // straight-through forward == hard
  }
}

extern "C" void kernel_launch(void* const* d_in, const int* in_sizes, int n_in,
                              void* d_out, int out_size, void* d_ws, size_t ws_size,
                              hipStream_t stream) {
  const float* x       = (const float*)d_in[0];   // (8,64,64,64) fp32
  const float* centers = (const float*)d_in[1];   // (16,) fp32
  float* out           = (float*)d_out;
  const int n          = in_sizes[0];             // 2,097,152

  const int block = 256;                          // 8 waves per block
  const int waves_per_block = block / 32;
  const int tiles = n / 32;
  int grid = (tiles + waves_per_block - 1) / waves_per_block;
  if (grid > 2048) grid = 2048;                   // ~16K waves, ~4 tiles/wave
  if (grid < 1) grid = 1;

  hipLaunchKernelGGL(softquant_wmma_kernel, dim3(grid), dim3(block), 0, stream,
                     x, centers, out, n);
}
